// GraphTransformerComm_52810917872098
// MI455X (gfx1250) — matrix operand order repaired and checked by reading, not compiled, not run on hardware
//
#include <hip/hip_runtime.h>
#include <hip/hip_bf16.h>

// ---------------------------------------------------------------------------
// GraphTransformerComm on MI455X (gfx1250, wave32, WMMA)
// All large GEMMs go through v_wmma_f32_16x16x32_bf16 (bf16 in, f32 acc).
// Pure-copy LDS staging uses global_load_async_to_lds_b128 (ASYNCcnt path).
// ---------------------------------------------------------------------------

#define NN   64
#define BB   256
#define CC   256
#define MM   128
#define HH   4
#define TT   8
#define TKP  8
#define HD   64

typedef unsigned int u32;
typedef __bf16 bf16;
typedef __attribute__((ext_vector_type(16))) __bf16 v16bf;
typedef __attribute__((ext_vector_type(8)))  float  v8f;
typedef __attribute__((ext_vector_type(8)))  u32    v8u;

#if defined(__gfx1250__)
#define USE_ASYNC_LDS 1
#else
#define USE_ASYNC_LDS 0
#endif

#if USE_ASYNC_LDS
// GLOBAL_LOAD_ASYNC_TO_LDS_B128: vdst = LDS byte offset (addr[31:0] of the
// shared-aperture flat pointer), vaddr = 64-bit global address. ASYNCcnt.
__device__ __forceinline__ void async_g2l_b128(void* lds_ptr, const void* gptr) {
    u32 lds_off = (u32)(uintptr_t)lds_ptr;
    asm volatile("global_load_async_to_lds_b128 %0, %1, off"
                 :: "v"(lds_off), "v"(gptr) : "memory");
}
__device__ __forceinline__ void wait_async0() {
    asm volatile("s_wait_asynccnt 0x0" ::: "memory");
}
#endif

__device__ __forceinline__ bf16 f2bf(float f) {
    u32 u = __builtin_bit_cast(u32, f);
    u32 r = (u + 0x7FFFu + ((u >> 16) & 1u)) >> 16;   // round-to-nearest-even
    unsigned short s = (unsigned short)r;
    return __builtin_bit_cast(bf16, s);
}

__device__ __forceinline__ v8f zero8() {
    v8f z;
#pragma unroll
    for (int i = 0; i < 8; ++i) z[i] = 0.0f;
    return z;
}

// A-fragment: 16xK row-major bf16 in LDS, ld = row stride (elements).
// Lane l holds row (rowbase + l&15); K split per ISA 7.12.2:
//   lanes 0-15: K = {0..7, 16..23}+k0 ; lanes 16-31: K = {8..15, 24..31}+k0
__device__ __forceinline__ v16bf load_a_frag(const bf16* base, int ld,
                                             int rowbase, int k0) {
    int lane = threadIdx.x & 31;
    int half = lane >> 4;
    const bf16* p = base + (rowbase + (lane & 15)) * ld + k0 + 8 * half;
    v8u t;
    t[0] = *(const u32*)(p + 0);
    t[1] = *(const u32*)(p + 2);
    t[2] = *(const u32*)(p + 4);
    t[3] = *(const u32*)(p + 6);
    t[4] = *(const u32*)(p + 16);
    t[5] = *(const u32*)(p + 18);
    t[6] = *(const u32*)(p + 20);
    t[7] = *(const u32*)(p + 22);
    return __builtin_bit_cast(v16bf, t);
}

// B-fragment: KxN row-major bf16 in LDS, ld = row stride (elements).
// Lane l holds B-column c0 + (l&15); K = {0..7, 16..23}+k0 (+8 for lanes 16-31).
__device__ __forceinline__ v16bf load_b_frag(const bf16* base, int ld,
                                             int k0, int c0) {
    int lane = threadIdx.x & 31;
    const bf16* p = base + (k0 + 8 * (lane >> 4)) * ld + c0 + (lane & 15);
    v16bf t;
#pragma unroll
    for (int s = 0; s < 16; ++s) t[s] = p[((s & 7) + ((s >> 3) << 4)) * ld];
    return t;
}

__device__ __forceinline__ v8f wmma_bf16(v16bf a, v16bf b, v8f c) {
    return __builtin_amdgcn_wmma_f32_16x16x32_bf16(
        false, a, false, b, (short)0, c, false, false);
}

// ---------------------------------------------------------------------------
// Kernel: roll buffers[:, :, 1:, :] -> new_buf[:, :, :T-1, :]
// ---------------------------------------------------------------------------
__global__ __launch_bounds__(256)
void k_roll(const float* __restrict__ buffers, float* __restrict__ nb) {
    size_t p = blockIdx.x;                       // (n,b) pair, N*B total
    const float* src = buffers + (p * TT + 1) * CC;
    float* dst = nb + p * TT * CC;
#pragma unroll
    for (int i = threadIdx.x; i < (TT - 1) * CC; i += 256) dst[i] = src[i];
}

// ---------------------------------------------------------------------------
// Kernel: per-(b,h) graph attention. 4 waves, 128 threads.
// ---------------------------------------------------------------------------
__global__ __launch_bounds__(128)
void k_graph_attn(const float* __restrict__ q, const float* __restrict__ k,
                  const float* __restrict__ v, const float* __restrict__ sg,
                  const float* __restrict__ ctrl, const float* __restrict__ bond,
                  float* __restrict__ attn_out, bf16* __restrict__ msgs_ws) {
    __shared__ __align__(32) bf16 Qs[NN][HD];
    __shared__ __align__(32) bf16 KsT[HD][NN];   // K transposed: [d][m]
    __shared__ __align__(32) bf16 Vs[NN][HD];    // Vg row-major: [m][d]
    __shared__ __align__(32) bf16 Abf[NN][NN];
    __shared__ float Ls[NN][NN + 4];
    __shared__ float cbias[NN];

    const int b  = blockIdx.x >> 2;
    const int hh = blockIdx.x & 3;
    const int tid = threadIdx.x;

    for (int idx = tid; idx < NN * HD; idx += 128) {
        int n = idx >> 6, d = idx & 63;
        size_t g = ((size_t)n * BB + b) * CC + (size_t)hh * HD + d;
        Qs[n][d]  = f2bf(q[g] * 0.125f);         // fold in 1/sqrt(HD)
        KsT[d][n] = f2bf(k[g]);
        Vs[n][d]  = f2bf(v[g] * sg[g]);
    }
    for (int m = tid; m < NN; m += 128)
        cbias[m] = ctrl[((size_t)m * BB + b) * HH + hh];
    __syncthreads();

    const int w = tid >> 5, lane = tid & 31, half = lane >> 4, r = lane & 15;

    // logits[n,m] = Q·K^T : A = Qs, B = KsT (row-major [d][m])
#pragma unroll
    for (int mi = 0; mi < 4; ++mi) {
        v8f acc = zero8();
#pragma unroll
        for (int k0 = 0; k0 < HD; k0 += 32) {
            v16bf af = load_a_frag(&Qs[0][0], HD, 16 * w, k0);
            v16bf bf = load_b_frag(&KsT[0][0], NN, k0, 16 * mi);
            acc = wmma_bf16(af, bf, acc);
        }
#pragma unroll
        for (int j = 0; j < 8; ++j) {
            int n = 16 * w + j + 8 * half;
            int m = 16 * mi + r;
            float val = acc[j] + bond[n * NN + m] + cbias[m];
            if (n == m) val = -3.0e38f;          // self mask
            Ls[n][m] = val;
        }
    }
    __syncthreads();

    // Per-row top-8 threshold + softmax (one thread per row, serial over 64)
    if (tid < NN) {
        int n = tid;
        float t8[TKP];
#pragma unroll
        for (int i = 0; i < TKP; ++i) t8[i] = -3.4e38f;
        for (int m = 0; m < NN; ++m) {
            float vv = Ls[n][m];
            if (vv > t8[TKP - 1]) {
                t8[TKP - 1] = vv;
#pragma unroll
                for (int i = TKP - 1; i > 0; --i)
                    if (t8[i] > t8[i - 1]) {
                        float tmp = t8[i]; t8[i] = t8[i - 1]; t8[i - 1] = tmp;
                    }
            }
        }
        float thr = t8[TKP - 1], mx = t8[0], sum = 0.0f;
        for (int m = 0; m < NN; ++m) {
            float vv = Ls[n][m];
            float e = (vv >= thr) ? __expf(vv - mx) : 0.0f;
            Ls[n][m] = e;
            sum += e;
        }
        float inv = 1.0f / sum;
        for (int m = 0; m < NN; ++m) Ls[n][m] *= inv;
    }
    __syncthreads();

    // Emit attn output + bf16 copy for the msgs GEMM
    size_t attn_base = (((size_t)b * HH + hh) * NN) * NN;
    for (int idx = tid; idx < NN * NN; idx += 128) {
        int n = idx >> 6, m = idx & 63;
        float a = Ls[n][m];
        attn_out[attn_base + idx] = a;
        Abf[n][m] = f2bf(a);
    }
    __syncthreads();

    // msgs[n,d] = attn · Vg : A = Abf, B = Vs ([m][d] row-major)
#pragma unroll
    for (int di = 0; di < 4; ++di) {
        v8f acc = zero8();
#pragma unroll
        for (int k0 = 0; k0 < NN; k0 += 32) {
            v16bf af = load_a_frag(&Abf[0][0], NN, 16 * w, k0);
            v16bf bf = load_b_frag(&Vs[0][0], HD, k0, 16 * di);
            acc = wmma_bf16(af, bf, acc);
        }
#pragma unroll
        for (int j = 0; j < 8; ++j) {
            int n = 16 * w + j + 8 * half;
            int d = 16 * di + r;
            msgs_ws[((size_t)b * NN + n) * CC + (size_t)hh * HD + d] = f2bf(acc[j]);
        }
    }
}

// ---------------------------------------------------------------------------
// Kernel: msgs @ out_w + out_b, recv gating -> new_buf[:, :, T-1, :]
// grid (B*N/64, C/64), 128 threads. A-tile staged via async global->LDS.
// ---------------------------------------------------------------------------
__global__ __launch_bounds__(128)
void k_out_proj(const bf16* __restrict__ msgs, const float* __restrict__ out_w,
                const float* __restrict__ out_b, const float* __restrict__ recv,
                float* __restrict__ nb) {
    __shared__ __align__(32) bf16 As[64][64];
    __shared__ __align__(32) bf16 Ws[64][64];
    const int r0 = blockIdx.x * 64, c0 = blockIdx.y * 64;
    const int tid = threadIdx.x, w = tid >> 5, lane = tid & 31;
    const int half = lane >> 4, r = lane & 15;

    v8f acc[4];
#pragma unroll
    for (int i = 0; i < 4; ++i) acc[i] = zero8();

    for (int kc = 0; kc < 4; ++kc) {
#if USE_ASYNC_LDS
        // A tile is already bf16 in the workspace: pure 8 KB copy -> async DMA
#pragma unroll
        for (int it = 0; it < 4; ++it) {
            int chunk = tid + 128 * it;           // 512 chunks x 16 B
            int row = chunk >> 3, seg = (chunk & 7) * 8;
            async_g2l_b128(&As[row][seg],
                           msgs + (size_t)(r0 + row) * CC + kc * 64 + seg);
        }
        for (int idx = tid; idx < 4096; idx += 128) {
            int i = idx >> 6, kk = idx & 63;
            Ws[i][kk] = f2bf(out_w[(size_t)(kc * 64 + i) * CC + c0 + kk]);
        }
        wait_async0();
#else
        for (int idx = tid; idx < 4096; idx += 128) {
            int i = idx >> 6, kk = idx & 63;
            As[i][kk] = msgs[(size_t)(r0 + i) * CC + kc * 64 + kk];
            Ws[i][kk] = f2bf(out_w[(size_t)(kc * 64 + i) * CC + c0 + kk]);
        }
#endif
        __syncthreads();
#pragma unroll
        for (int ci = 0; ci < 4; ++ci)
#pragma unroll
            for (int k0 = 0; k0 < 64; k0 += 32) {
                v16bf af = load_a_frag(&As[0][0], 64, 16 * w, k0);
                v16bf bf = load_b_frag(&Ws[0][0], 64, k0, 16 * ci);
                acc[ci] = wmma_bf16(af, bf, acc[ci]);
            }
        __syncthreads();
    }

#pragma unroll
    for (int ci = 0; ci < 4; ++ci)
#pragma unroll
        for (int j = 0; j < 8; ++j) {
            int gr = r0 + 16 * w + j + 8 * half;  // row = b*N + n
            int c  = c0 + 16 * ci + r;
            int bb = gr >> 6, n = gr & 63;
            float val = acc[ci][j] + out_b[c];
            val *= recv[((size_t)n * BB + bb) * CC + c];
            nb[(((size_t)n * BB + bb) * TT + (TT - 1)) * CC + c] = val;
        }
}

// ---------------------------------------------------------------------------
// Kernel: bq = h @ bq_w + bq_b   (rows = N*B, K = M = 128)
// ---------------------------------------------------------------------------
__global__ __launch_bounds__(128)
void k_bq(const float* __restrict__ h, const float* __restrict__ bq_w,
          const float* __restrict__ bq_b, float* __restrict__ bq_out) {
    __shared__ __align__(32) bf16 As[64][64];
    __shared__ __align__(32) bf16 Ws[64][64];
    const int r0 = blockIdx.x * 64, c0 = blockIdx.y * 64;
    const int tid = threadIdx.x, w = tid >> 5, lane = tid & 31;
    const int half = lane >> 4, r = lane & 15;

    v8f acc[4];
#pragma unroll
    for (int i = 0; i < 4; ++i) acc[i] = zero8();

    for (int kc = 0; kc < 2; ++kc) {              // K = 128
        for (int idx = tid; idx < 4096; idx += 128) {
            int i = idx >> 6, kk = idx & 63;
            As[i][kk] = f2bf(h[(size_t)(r0 + i) * MM + kc * 64 + kk]);
            Ws[i][kk] = f2bf(bq_w[(size_t)(kc * 64 + i) * CC + c0 + kk]);
        }
        __syncthreads();
#pragma unroll
        for (int ci = 0; ci < 4; ++ci)
#pragma unroll
            for (int k0 = 0; k0 < 64; k0 += 32) {
                v16bf af = load_a_frag(&As[0][0], 64, 16 * w, k0);
                v16bf bf = load_b_frag(&Ws[0][0], 64, k0, 16 * ci);
                acc[ci] = wmma_bf16(af, bf, acc[ci]);
            }
        __syncthreads();
    }

#pragma unroll
    for (int ci = 0; ci < 4; ++ci)
#pragma unroll
        for (int j = 0; j < 8; ++j) {
            int gr = r0 + 16 * w + j + 8 * half;
            int c  = c0 + 16 * ci + r;
            bq_out[(size_t)gr * CC + c] = acc[ci][j] + bq_b[c];
        }
}

// ---------------------------------------------------------------------------
// Kernel: fused temporal attention. One block = 8 (n,b) pairs = 64 buf rows.
// bk/bv GEMMs via WMMA; dot(bq,bk) via ds_add_f32; softmax over T; readout.
// bq staged via async global->LDS; next weight chunk prefetched.
// ---------------------------------------------------------------------------
__global__ __launch_bounds__(128)
void k_temporal(const float* __restrict__ nb, const float* __restrict__ bq,
                const float* __restrict__ bk_w, const float* __restrict__ bk_b,
                const float* __restrict__ bv_w, const float* __restrict__ bv_b,
                const float* __restrict__ decay_logit,
                float* __restrict__ readout) {
    __shared__ __align__(32) bf16 As[64][CC];     // 32 KB: 64 buf rows, K=256
    __shared__ __align__(32) bf16 Wk[64][64];     // 8 KB
    __shared__ __align__(32) bf16 Wv[64][64];     // 8 KB
    __shared__ __align__(16) float bvs[64][CC];   // 64 KB: decay-scaled bv
    __shared__ __align__(16) float bqs[8][CC];    // 8 KB
    __shared__ float dotv[64];
    __shared__ float bav[64];

    const size_t p0 = (size_t)blockIdx.x * 8;     // first (n,b) pair
    const int tid = threadIdx.x, w = tid >> 5, lane = tid & 31;
    const int half = lane >> 4, r = lane & 15;

#if USE_ASYNC_LDS
    // bq rows p0..p0+7 are one flat contiguous 8 KB region: async DMA to LDS
#pragma unroll
    for (int it = 0; it < 4; ++it) {
        int chunk = tid + 128 * it;               // 512 chunks x 16 B
        async_g2l_b128((char*)&bqs[0][0] + (size_t)chunk * 16,
                       bq + p0 * CC + (size_t)chunk * 4);
    }
#else
    for (int idx = tid; idx < 8 * CC; idx += 128)
        bqs[idx >> 8][idx & 255] = bq[p0 * CC + idx];
#endif

    const float* arow = nb + p0 * TT * CC;        // 64 contiguous rows
    for (int idx = tid; idx < 64 * CC; idx += 128)
        As[idx >> 8][idx & 255] = f2bf(arow[idx]);
    if (tid < 64) dotv[tid] = 0.0f;
    const float decay = 1.0f / (1.0f + __expf(-decay_logit[0]));
#if USE_ASYNC_LDS
    wait_async0();
#endif
    __syncthreads();

    for (int cb = 0; cb < 4; ++cb) {
        const int c0 = cb * 64;
        v8f acck[4], accv[4];
#pragma unroll
        for (int i = 0; i < 4; ++i) { acck[i] = zero8(); accv[i] = zero8(); }

        for (int kc = 0; kc < 4; ++kc) {
            for (int idx = tid; idx < 4096; idx += 128) {
                int kk = idx >> 6, cc = idx & 63;
                Wk[kk][cc] = f2bf(bk_w[(size_t)(kc * 64 + kk) * CC + c0 + cc]);
                Wv[kk][cc] = f2bf(bv_w[(size_t)(kc * 64 + kk) * CC + c0 + cc]);
            }
            if (kc < 3) {                          // warm L2/WGP$ for next chunk
                size_t nofs = (size_t)((kc + 1) * 64 + (tid >> 1)) * CC +
                              c0 + (tid & 1) * 32;
                __builtin_prefetch(bk_w + nofs, 0, 1);
                __builtin_prefetch(bv_w + nofs, 0, 1);
            }
            __syncthreads();
#pragma unroll
            for (int ci = 0; ci < 4; ++ci)
#pragma unroll
                for (int k0 = 0; k0 < 64; k0 += 32) {
                    v16bf af = load_a_frag(&As[0][0], CC, 16 * w, kc * 64 + k0);
                    v16bf fk = load_b_frag(&Wk[0][0], 64, k0, 16 * ci);
                    acck[ci] = wmma_bf16(af, fk, acck[ci]);
                    v16bf fv = load_b_frag(&Wv[0][0], 64, k0, 16 * ci);
                    accv[ci] = wmma_bf16(af, fv, accv[ci]);
                }
            __syncthreads();
        }

        // epilogue for this 64-column slab
#pragma unroll
        for (int j = 0; j < 8; ++j) {
            int row = 16 * w + j + 8 * half;      // 0..63 ; pair = row>>3, t = row&7
            int t = row & 7;
            float tw = __powf(decay, (float)(TT - t));
            float dsum = 0.0f;
#pragma unroll
            for (int ci = 0; ci < 4; ++ci) {
                int c = c0 + 16 * ci + r;
                float bkv = acck[ci][j] + bk_b[c];
                dsum += bkv * bqs[row >> 3][c];
                float bvv = (accv[ci][j] + bv_b[c]) * tw;
                bvs[row][c] = bvv;
            }
            atomicAdd(&dotv[row], dsum);          // ds_add_f32
        }
    }
    __syncthreads();

    // softmax over T per pair, scale 1/sqrt(C) = 1/16
    if (tid < 8) {
        int l = tid;
        float mx = -3.4e38f;
        for (int t = 0; t < TT; ++t) mx = fmaxf(mx, dotv[l * 8 + t]);
        float s = 0.0f;
        for (int t = 0; t < TT; ++t) {
            float e = __expf((dotv[l * 8 + t] - mx) * 0.0625f);
            bav[l * 8 + t] = e;
            s += e;
        }
        float inv = 1.0f / s;
        for (int t = 0; t < TT; ++t) bav[l * 8 + t] *= inv;
    }
    __syncthreads();

    // readout[p, c] = sum_t ba[t] * bv[t, c]
    for (int idx = tid; idx < 8 * CC; idx += 128) {
        int l = idx >> 8, c = idx & 255;
        float s = 0.0f;
#pragma unroll
        for (int t = 0; t < TT; ++t) s += bav[l * 8 + t] * bvs[l * 8 + t][c];
        readout[(p0 + l) * CC + c] = s;
    }
}

// ---------------------------------------------------------------------------
extern "C" void kernel_launch(void* const* d_in, const int* in_sizes, int n_in,
                              void* d_out, int out_size, void* d_ws, size_t ws_size,
                              hipStream_t stream) {
    const float* q    = (const float*)d_in[0];
    const float* k    = (const float*)d_in[1];
    const float* v    = (const float*)d_in[2];
    const float* sg   = (const float*)d_in[3];
    const float* recv = (const float*)d_in[4];
    const float* ctrl = (const float*)d_in[5];
    const float* h    = (const float*)d_in[6];
    const float* buf  = (const float*)d_in[7];
    const float* bond = (const float*)d_in[8];
    const float* out_w = (const float*)d_in[9];
    const float* out_b = (const float*)d_in[10];
    const float* bq_w  = (const float*)d_in[11];
    const float* bq_b  = (const float*)d_in[12];
    const float* bk_w  = (const float*)d_in[13];
    const float* bk_b  = (const float*)d_in[14];
    const float* bv_w  = (const float*)d_in[15];
    const float* bv_b  = (const float*)d_in[16];
    const float* dlog  = (const float*)d_in[17];

    float* out  = (float*)d_out;
    float* ro   = out;                                       // [N,B,C]
    float* nb   = out + (size_t)NN * BB * CC;                // [N,B,T,C]
    float* attn = nb + (size_t)NN * BB * TT * CC;            // [B,H,N,N]

    bf16*  ws_msgs = (bf16*)d_ws;                            // [B,N,C] bf16
    float* ws_bq   = (float*)((char*)d_ws + (size_t)BB * NN * CC * sizeof(bf16));

    // 1) roll old buffers into new_buf
    k_roll<<<NN * BB, 256, 0, stream>>>(buf, nb);
    // 2) graph attention per (b,h): attn output + bf16 msgs
    k_graph_attn<<<BB * HH, 128, 0, stream>>>(q, k, v, sg, ctrl, bond, attn, ws_msgs);
    // 3) output projection + recv gate -> new_buf last slot
    k_out_proj<<<dim3(BB * NN / 64, CC / 64), 128, 0, stream>>>(ws_msgs, out_w, out_b, recv, nb);
    // 4) bq = h @ bq_w + bq_b
    k_bq<<<dim3(NN * BB / 64, CC / 64), 128, 0, stream>>>(h, bq_w, bq_b, ws_bq);
    // 5) fused temporal bk/bv + softmax + readout
    k_temporal<<<NN * BB / 8, 128, 0, stream>>>(nb, ws_bq, bk_w, bk_b, bv_w, bv_b, dlog, ro);
}
